// Model_30262339567623
// MI455X (gfx1250) — matrix operand order repaired and checked by reading, not compiled, not run on hardware
//
#include <hip/hip_runtime.h>

// ---------------------------------------------------------------------------
// V=10000, R=8192, D=1024, B=1024, L=64, M=65536.
//   RNN:   64x  H_t = tanh([x_t | H_{t-1}] @ [Wxh;Whh] + bh)   (bf16 WMMA)
//   Score: gathered [M,2048] @ W1 bf16 WMMA GEMM, fused +b1 -> tanh ->
//          dot(W2[:,r]) epilogue (LDS ds_add_f32 reduction + one global
//          atomicAdd per row).
// Weights are packed TRANSPOSED ([N][K] bf16) so the WMMA B-operand layout
// (lane = column, per-lane K in two 8-elem runs) is serviced by two 16B LDS
// reads. The Tensor Data Mover DMAs the B tile into LDS with pad_enable
// (16-dword interval + 4-dword pad -> 80B row stride: 16B-aligned b128 reads,
// conflict-free banks). Double-buffered LDS pipeline: next tile's A staging +
// TDM issue overlap the current tile's 8 WMMAs per wave (block 128x128).
// tanh uses gfx1250 hardware v_tanh_f32. Workspace ~160 MB.
// ---------------------------------------------------------------------------

typedef __attribute__((ext_vector_type(16))) __bf16 v16bf;
typedef __attribute__((ext_vector_type(8)))  float  v8f;
typedef __attribute__((ext_vector_type(4)))  unsigned v4u;
typedef __attribute__((ext_vector_type(8)))  int      v8i;
typedef __attribute__((ext_vector_type(4)))  int      v4i;

union Frag16 { v16bf v; unsigned u[8]; };

#if defined(__gfx1250__) && __has_builtin(__builtin_amdgcn_tensor_load_to_lds)
#define USE_TDM 1
#else
#define USE_TDM 0
#endif

__device__ __forceinline__ float fast_tanh(float x) {
#if __has_builtin(__builtin_amdgcn_tanhf)
  return __builtin_amdgcn_tanhf(x);
#elif __has_builtin(__builtin_amdgcn_tanh_f32)
  return __builtin_amdgcn_tanh_f32(x);
#else
  return tanhf(x);
#endif
}

__device__ __forceinline__ unsigned short f32_to_bf16(float f) {
  unsigned u = __float_as_uint(f);
  unsigned r = u + 0x7FFFu + ((u >> 16) & 1u);  // round-to-nearest-even
  return (unsigned short)(r >> 16);
}

#define D_DIM   1024
#define K2_DIM  2048
#define L_DIM   64
#define R_DIM   8192
#define BM      128
#define BN      128
#define BK      32
#define TSTR    40    // ushort row stride for LDS tiles (80B: 16B-aligned,
                      // 20 dwords -> 16 distinct banks across a 16-row frag)
#define NIT     (K2_DIM / BK)

// TDM group1 flags: data_size=1(2B) | pad_enable | pad_interval=3 (16 dwords
// = one 32-elem bf16 tile row) | pad_amount=3 (4 dwords) -> LDS stride TSTR.
#define TDM_G1_FLAGS 0x06D10000u

// --------------------------- packing kernels --------------------------------

__global__ __launch_bounds__(256) void k_f32_to_bf16(
    const float* __restrict__ s, unsigned short* __restrict__ d, int n) {
  int i = blockIdx.x * 256 + threadIdx.x;
  if (i < n) d[i] = f32_to_bf16(s[i]);
}

// Transpose-pack: s is row-major [K x N] f32; write d[n * dstride + coff + k].
__global__ __launch_bounds__(256) void k_pack_T(
    const float* __restrict__ s, unsigned short* __restrict__ d,
    int K, int N, int dstride, int coff) {
  int i = blockIdx.x * 256 + threadIdx.x;
  if (i < K * N) {
    int k = i / N, n = i - k * N;
    d[(size_t)n * dstride + coff + k] = f32_to_bf16(s[i]);
  }
}

__global__ __launch_bounds__(256) void k_init_h(
    const float* __restrict__ h0, unsigned short* __restrict__ H, int n) {
  int i = blockIdx.x * 256 + threadIdx.x;
  if (i < n) H[i] = f32_to_bf16(h0[i & (D_DIM - 1)]);
}

// ----------------------------- TDM tile loader ------------------------------
// D# per CDNA5 ISA §8.3-8.5. 2D tile (tile_dim1 rows x tile_dim0 elems) of a
// row-major [tensor_dim1 x tensor_dim0] bf16 matrix -> LDS (with padding).

#if USE_TDM
__device__ __forceinline__ void tdm_load_tile_2d(
    const unsigned short* gsrc, unsigned lds_off,
    unsigned tile_dim0, unsigned tile_dim1,
    unsigned tensor_dim0, unsigned tensor_dim1, unsigned dim0_stride,
    unsigned flags) {
  unsigned long long ga = (unsigned long long)(uintptr_t)gsrc;
  v4u g0;
  g0[0] = 1u;                                   // count=1 (valid), user mode
  g0[1] = lds_off;                              // lds_addr (bytes)
  g0[2] = (unsigned)(ga & 0xFFFFFFFFull);       // global_addr[31:0]
  g0[3] = (unsigned)((ga >> 32) & 0x01FFFFFFull) | (2u << 30);  // hi | type=2
  v8i g1;
  g1[0] = (int)flags;                           // data_size / pad controls
  g1[1] = (int)(tensor_dim0 << 16);             // tensor_dim0[15:0] @63:48
  g1[2] = (int)((tensor_dim0 >> 16) | (tensor_dim1 << 16));
  g1[3] = (int)((tensor_dim1 >> 16) | (tile_dim0 << 16));
  g1[4] = (int)(tile_dim1 & 0xFFFFu);           // tile_dim1 (tile_dim2=0)
  g1[5] = (int)dim0_stride;                     // tensor_dim0_stride[31:0]
  g1[6] = 0;
  g1[7] = 0;
  v4i z4 = {0, 0, 0, 0};
#if __clang_major__ >= 23
  v8i z8 = {0, 0, 0, 0, 0, 0, 0, 0};
  __builtin_amdgcn_tensor_load_to_lds(g0, g1, z4, z4, z8, 0);
#else
  __builtin_amdgcn_tensor_load_to_lds(g0, g1, z4, z4, 0);
#endif
}
#endif

// ------------------------- fragment loaders (ISA 7.12.2) --------------------
// A (16x32): lane l(0-15)=row M, hi selects K 8-15/24-31; dword v = K pair.
__device__ __forceinline__ v16bf load_a_frag(const unsigned short* row0,
                                             int l, int hi) {
  Frag16 f;
  const unsigned* p = (const unsigned*)(row0 + l * TSTR);
#pragma unroll
  for (int v = 0; v < 8; ++v)
    f.u[v] = p[8 * (v >> 2) + 4 * hi + (v & 3)];
  return f.v;
}

// B (32x16): lane l = column n (tile stored as [n][k] rows); per-lane K
// matches A: hi selects K 8-15/24-31; dword v = K pair.
__device__ __forceinline__ v16bf load_b_frag(const unsigned short* row0,
                                             int l, int hi) {
  Frag16 f;
  const unsigned* p = (const unsigned*)(row0 + l * TSTR);
#pragma unroll
  for (int v = 0; v < 8; ++v)
    f.u[v] = p[8 * (v >> 2) + 4 * hi + (v & 3)];
  return f.v;
}

// ------------------------------ RNN step kernel -----------------------------
// H_out[1024,1024] = tanh( [emb[tok_t] | H_in] @ WcT^T + bh )
// grid (8, 8), 256 threads = 8 waves, wave tile 32x64 (8 WMMAs / K-step).

__global__ __launch_bounds__(256) void k_rnn_step(
    const int* __restrict__ tokens,
    const unsigned short* __restrict__ embb,
    const unsigned short* __restrict__ Hin,
    unsigned short* __restrict__ Hout,
    const unsigned short* __restrict__ WcT,   // [1024][2048] bf16 (transposed)
    const float* __restrict__ bh,
    unsigned short* __restrict__ enc,
    int t) {
  __shared__ __align__(16) unsigned short sA[2][BM * TSTR];
  __shared__ __align__(16) unsigned short sB[2][BN * TSTR];
  __shared__ int stok[BM];

  const int tid = threadIdx.x;
  const int m0 = blockIdx.x * BM;
  const int n0 = blockIdx.y * BN;
  if (tid < BM) stok[tid] = tokens[(m0 + tid) * L_DIM + t];
  __syncthreads();

  const int lane = tid & 31, wid = tid >> 5;
  const int wm = wid >> 1, wn = wid & 1;
  const int l = lane & 15, hi = lane >> 4;
#if USE_TDM
  const unsigned ldsB0 = (unsigned)(unsigned long long)(uintptr_t)(void*)&sB[0][0];
#endif

  auto stageA = [&](int k0, unsigned short* dst) {
    int row = tid >> 1, kc = (tid & 1) * 16;
    const unsigned short* src =
        (k0 < D_DIM) ? embb + (size_t)stok[row] * D_DIM + k0 + kc
                     : Hin + (size_t)(m0 + row) * D_DIM + (k0 - D_DIM) + kc;
    *(uint4*)&dst[row * TSTR + kc] = *(const uint4*)src;
    *(uint4*)&dst[row * TSTR + kc + 8] = *(const uint4*)(src + 8);
  };
#if USE_TDM
  auto stageB = [&](int k0, int buf) {
    tdm_load_tile_2d(WcT + (size_t)n0 * K2_DIM + k0,
                     ldsB0 + (unsigned)buf * (BN * TSTR * 2),
                     BK, BN, K2_DIM, D_DIM, K2_DIM, TDM_G1_FLAGS);
  };
#else
  auto stageB = [&](int k0, int buf) {  // straight row copy of WcT tile
    int row = tid >> 1, kc = (tid & 1) * 16;
    const unsigned short* src = WcT + (size_t)(n0 + row) * K2_DIM + k0 + kc;
    *(uint4*)&sB[buf][row * TSTR + kc] = *(const uint4*)src;
    *(uint4*)&sB[buf][row * TSTR + kc + 8] = *(const uint4*)(src + 8);
  };
#endif

  v8f acc[2][4];
#pragma unroll
  for (int mi = 0; mi < 2; ++mi)
#pragma unroll
    for (int j = 0; j < 4; ++j)
#pragma unroll
      for (int e = 0; e < 8; ++e) acc[mi][j][e] = 0.0f;

  // prologue: stage tile 0 into buffer 0
  stageA(0, sA[0]);
  if (USE_TDM ? (wid == 0) : true) stageB(0, 0);

  for (int i = 0; i < NIT; ++i) {
    int cur = i & 1;
#if USE_TDM
    if (wid == 0) __builtin_amdgcn_s_wait_tensorcnt(0);
#endif
    __syncthreads();  // tile i ready; tile i-1 fully consumed
    if (i + 1 < NIT) {
      stageA((i + 1) * BK, sA[cur ^ 1]);
      if (USE_TDM ? (wid == 0) : true) stageB((i + 1) * BK, cur ^ 1);
    }
    v16bf af0 = load_a_frag(&sA[cur][(wm * 32) * TSTR], l, hi);
    v16bf af1 = load_a_frag(&sA[cur][(wm * 32 + 16) * TSTR], l, hi);
#pragma unroll
    for (int j = 0; j < 4; ++j) {
      v16bf bf_ = load_b_frag(&sB[cur][(wn * 64 + j * 16) * TSTR], l, hi);
      acc[0][j] = __builtin_amdgcn_wmma_f32_16x16x32_bf16(
          false, af0, false, bf_, (short)0, acc[0][j], false, false);
      acc[1][j] = __builtin_amdgcn_wmma_f32_16x16x32_bf16(
          false, af1, false, bf_, (short)0, acc[1][j], false, false);
    }
  }

  // epilogue: + bh, tanh (v_tanh_f32), store bf16 into H ping-pong and enc
#pragma unroll
  for (int j = 0; j < 4; ++j) {
    int n = n0 + wn * 64 + j * 16 + l;
    float bias = bh[n];
#pragma unroll
    for (int mi = 0; mi < 2; ++mi)
#pragma unroll
      for (int r = 0; r < 8; ++r) {
        int m = m0 + wm * 32 + mi * 16 + r + 8 * hi;
        unsigned short hb = f32_to_bf16(fast_tanh(acc[mi][j][r] + bias));
        Hout[(size_t)m * D_DIM + n] = hb;
        enc[((size_t)m * L_DIM + t) * D_DIM + n] = hb;
      }
  }
}

// ------------------------------- score kernel -------------------------------
// h = tanh(cat(enc[i],enc[j]) @ W1T^T + b1); partial score over the block's
// 128 N-columns via LDS ds_add_f32, one global atomicAdd per row.
// grid (M/BM, 1024/BN) = (512, 8).

__global__ __launch_bounds__(256) void k_score(
    const unsigned short* __restrict__ enc,
    const int* __restrict__ i_idx,
    const int* __restrict__ j_idx,
    const int* __restrict__ r_idx,
    const unsigned short* __restrict__ W1T,   // [1024][2048] bf16 (transposed)
    const float* __restrict__ b1,
    const float* __restrict__ W2,
    const float* __restrict__ b2,
    float* __restrict__ out) {
  __shared__ __align__(16) unsigned short sA[2][BM * TSTR];
  __shared__ __align__(16) unsigned short sB[2][BN * TSTR];
  __shared__ int si[BM], sj[BM], sr[BM];
  __shared__ float ssc[BM];

  const int tid = threadIdx.x;
  const int m0 = blockIdx.x * BM;
  const int n0 = blockIdx.y * BN;
  if (tid < BM) {
    si[tid] = i_idx[m0 + tid];
    sj[tid] = j_idx[m0 + tid];
    sr[tid] = r_idx[m0 + tid];
    ssc[tid] = 0.0f;
  }
  __syncthreads();

  const int lane = tid & 31, wid = tid >> 5;
  const int wm = wid >> 1, wn = wid & 1;
  const int l = lane & 15, hi = lane >> 4;
#if USE_TDM
  const unsigned ldsB0 = (unsigned)(unsigned long long)(uintptr_t)(void*)&sB[0][0];
#endif

  auto stageA = [&](int k0, unsigned short* dst) {
    int row = tid >> 1, kc = (tid & 1) * 16;
    int srcrow = (k0 < D_DIM) ? si[row] : sj[row];
    int kk = (k0 < D_DIM) ? (k0 + kc) : (k0 - D_DIM + kc);
    const unsigned short* src = enc + (size_t)srcrow * D_DIM + kk;
    *(uint4*)&dst[row * TSTR + kc] = *(const uint4*)src;
    *(uint4*)&dst[row * TSTR + kc + 8] = *(const uint4*)(src + 8);
  };
#if USE_TDM
  auto stageB = [&](int k0, int buf) {
    tdm_load_tile_2d(W1T + (size_t)n0 * K2_DIM + k0,
                     ldsB0 + (unsigned)buf * (BN * TSTR * 2),
                     BK, BN, K2_DIM, D_DIM, K2_DIM, TDM_G1_FLAGS);
  };
#else
  auto stageB = [&](int k0, int buf) {
    int row = tid >> 1, kc = (tid & 1) * 16;
    const unsigned short* src = W1T + (size_t)(n0 + row) * K2_DIM + k0 + kc;
    *(uint4*)&sB[buf][row * TSTR + kc] = *(const uint4*)src;
    *(uint4*)&sB[buf][row * TSTR + kc + 8] = *(const uint4*)(src + 8);
  };
#endif

  v8f acc[2][4];
#pragma unroll
  for (int mi = 0; mi < 2; ++mi)
#pragma unroll
    for (int j = 0; j < 4; ++j)
#pragma unroll
      for (int e = 0; e < 8; ++e) acc[mi][j][e] = 0.0f;

  stageA(0, sA[0]);
  if (USE_TDM ? (wid == 0) : true) stageB(0, 0);

  for (int i = 0; i < NIT; ++i) {
    int cur = i & 1;
#if USE_TDM
    if (wid == 0) __builtin_amdgcn_s_wait_tensorcnt(0);
#endif
    __syncthreads();
    if (i + 1 < NIT) {
      stageA((i + 1) * BK, sA[cur ^ 1]);
      if (USE_TDM ? (wid == 0) : true) stageB((i + 1) * BK, cur ^ 1);
    }
    v16bf af0 = load_a_frag(&sA[cur][(wm * 32) * TSTR], l, hi);
    v16bf af1 = load_a_frag(&sA[cur][(wm * 32 + 16) * TSTR], l, hi);
#pragma unroll
    for (int j = 0; j < 4; ++j) {
      v16bf bf_ = load_b_frag(&sB[cur][(wn * 64 + j * 16) * TSTR], l, hi);
      acc[0][j] = __builtin_amdgcn_wmma_f32_16x16x32_bf16(
          false, af0, false, bf_, (short)0, acc[0][j], false, false);
      acc[1][j] = __builtin_amdgcn_wmma_f32_16x16x32_bf16(
          false, af1, false, bf_, (short)0, acc[1][j], false, false);
    }
  }

  // fused epilogue: tanh(+b1) * W2[n, r]  ->  LDS per-row partial
#pragma unroll
  for (int j = 0; j < 4; ++j) {
    int n = n0 + wn * 64 + j * 16 + l;
    float b1n = b1[n];
#pragma unroll
    for (int mi = 0; mi < 2; ++mi)
#pragma unroll
      for (int r = 0; r < 8; ++r) {
        int mloc = wm * 32 + mi * 16 + r + 8 * hi;
        float hv = fast_tanh(acc[mi][j][r] + b1n);
        float w2v = W2[(size_t)n * R_DIM + sr[mloc]];
        atomicAdd(&ssc[mloc], hv * w2v);
      }
  }
  __syncthreads();
  if (tid < BM) {
    float v = ssc[tid];
    if (blockIdx.y == 0) v += b2[sr[tid]];
    atomicAdd(&out[m0 + tid], v);
  }
}

// --------------------------------- launch -----------------------------------

extern "C" void kernel_launch(void* const* d_in, const int* in_sizes, int n_in,
                              void* d_out, int out_size, void* d_ws,
                              size_t ws_size, hipStream_t stream) {
  const int*   tokens = (const int*)d_in[0];
  const int*   i_idx  = (const int*)d_in[1];
  const int*   j_idx  = (const int*)d_in[2];
  const int*   r_idx  = (const int*)d_in[3];
  const float* emb    = (const float*)d_in[4];
  const float* Wxh    = (const float*)d_in[5];
  const float* Whh    = (const float*)d_in[6];
  const float* bh     = (const float*)d_in[7];
  const float* h0     = (const float*)d_in[8];
  const float* W1     = (const float*)d_in[9];
  const float* b1     = (const float*)d_in[10];
  const float* W2     = (const float*)d_in[11];
  const float* b2     = (const float*)d_in[12];

  // workspace carve-up (~160 MB total)
  char* p = (char*)d_ws;
  unsigned short* enc  = (unsigned short*)p; p += (size_t)65536 * 1024 * 2; // 128 MB
  unsigned short* embb = (unsigned short*)p; p += (size_t)10000 * 1024 * 2; //  20 MB
  unsigned short* WcT  = (unsigned short*)p; p += (size_t)1024 * 2048 * 2;  //   4 MB
  unsigned short* W1T  = (unsigned short*)p; p += (size_t)1024 * 2048 * 2;  //   4 MB
  unsigned short* Ha   = (unsigned short*)p; p += (size_t)1024 * 1024 * 2;  //   2 MB
  unsigned short* Hb   = (unsigned short*)p; p += (size_t)1024 * 1024 * 2;  //   2 MB

  hipMemsetAsync(d_out, 0, (size_t)65536 * sizeof(float), stream);

  int n;
  n = 10000 * 1024;
  k_f32_to_bf16<<<(n + 255) / 256, 256, 0, stream>>>(emb, embb, n);
  // transposed weight packs: WcT[n][k] = {Wxh,k<1024 ; Whh,k>=1024}[k][n]
  n = 1024 * 1024;
  k_pack_T<<<(n + 255) / 256, 256, 0, stream>>>(Wxh, WcT, 1024, 1024, 2048, 0);
  k_pack_T<<<(n + 255) / 256, 256, 0, stream>>>(Whh, WcT, 1024, 1024, 2048, 1024);
  n = 2048 * 1024;
  k_pack_T<<<(n + 255) / 256, 256, 0, stream>>>(W1, W1T, 2048, 1024, 2048, 0);
  n = 1024 * 1024;
  k_init_h<<<(n + 255) / 256, 256, 0, stream>>>(h0, Ha, n);

  unsigned short* Hin = Ha;
  unsigned short* Hout = Hb;
  for (int t = 0; t < 64; ++t) {
    k_rnn_step<<<dim3(8, 8), 256, 0, stream>>>(tokens, embb, Hin, Hout, WcT,
                                               bh, enc, t);
    unsigned short* tmp = Hin; Hin = Hout; Hout = tmp;
  }

  k_score<<<dim3(512, 8), 256, 0, stream>>>(enc, i_idx, j_idx, r_idx, W1T,
                                            b1, W2, b2, (float*)d_out);
}
